// MultiScaleDeformableCrossAttentionAlignment_69054484185363
// MI455X (gfx1250) — matrix operand order repaired and checked by reading, not compiled, not run on hardware
//
#include <hip/hip_runtime.h>

typedef __attribute__((ext_vector_type(16))) _Float16 v16h;
typedef __attribute__((ext_vector_type(8)))  float    v8f;
typedef __attribute__((ext_vector_type(4)))  _Float16 h4;

#define BM 128
#define BN 64
#define BK 32
#define LDP 40   // LDS row pitch in halfs: 80B rows keep every fragment slice 16B-aligned

// ---- WMMA fragment gathers, contiguous k = 16*(lane>>4) + e in A and B (16-bit A 16x32, B 32x16, f32 C/D) ----
__device__ __forceinline__ v16h frag_a(const _Float16* As, int mbase, int lane) {
  v16h f;
  const _Float16* row = As + (mbase + (lane & 15)) * LDP;
  const int kb = (lane & 16) ? 16 : 0;
#pragma unroll
  for (int e = 0; e < 8; ++e) f[e] = row[kb + e];
#pragma unroll
  for (int e = 0; e < 8; ++e) f[e + 8] = row[kb + 8 + e];
  return f;
}

__device__ __forceinline__ v16h frag_b(const _Float16* Bs, int nbase, int lane) {
  v16h f;
  const _Float16* row = Bs + (nbase + (lane & 15)) * LDP;
  const int kb = (lane & 16) ? 16 : 0;
#pragma unroll
  for (int e = 0; e < 16; ++e) f[e] = row[kb + e];
  return f;
}

// D[z,m,n] = sum_k A[z,m,k]*B[n,k] + bias[n]
// AKC: A k-axis contiguous (sa_k==1), else m-axis contiguous (sa_m==1, e.g. res maps)
// BKC: B k-axis contiguous (sb_k==1), else n-axis contiguous (sb_n==1)
// All shapes are exact multiples of (BM,BN,BK): no edge guards.
template <bool AKC, bool BKC, bool OUTH>
__global__ void gemm_wmma_f16(
    const float* __restrict__ A, long long sa_m, long long sa_k, long long a_batch,
    const float* __restrict__ B, long long sb_n, long long sb_k,
    const float* __restrict__ bias,
    float* __restrict__ Df, _Float16* __restrict__ Dh,
    long long d_batch, int N, int K)
{
  __shared__ _Float16 As[BM * LDP];
  __shared__ _Float16 Bs[BN * LDP];

  const int t    = threadIdx.x;
  const int lane = t & 31;
  const int wave = t >> 5;
  const int wm   = (wave & 3) * 32;   // 4 waves along M
  const int wn   = (wave >> 2) * 32;  // 2 waves along N
  const int bm   = blockIdx.x * BM;
  const int bn   = blockIdx.y * BN;
  const long long z = blockIdx.z;

  const float* __restrict__ Az = A + z * a_batch;

  float4 pa[4];   // 16 A elements / thread
  float4 pb[2];   //  8 B elements / thread

  auto load_tiles = [&](int k0) {
#pragma unroll
    for (int it = 0; it < 4; ++it) {
      const int idx = t + (it << 8);            // 0..1023 float4s of A tile
      if constexpr (AKC) {
        const int row = idx >> 3, k4 = (idx & 7) << 2;
        pa[it] = *(const float4*)(Az + (long long)(bm + row) * sa_m + (k0 + k4));
      } else {
        const int col = idx >> 5, r4 = (idx & 31) << 2;
        pa[it] = *(const float4*)(Az + (long long)(k0 + col) * sa_k + (bm + r4));
      }
    }
#pragma unroll
    for (int it = 0; it < 2; ++it) {
      const int idx = t + (it << 8);            // 0..511 float4s of B tile
      if constexpr (BKC) {
        const int row = idx >> 3, k4 = (idx & 7) << 2;
        pb[it] = *(const float4*)(B + (long long)(bn + row) * sb_n + (k0 + k4));
      } else {
        const int col = idx >> 4, n4 = (idx & 15) << 2;
        pb[it] = *(const float4*)(B + (long long)(k0 + col) * sb_k + (bn + n4));
      }
    }
  };

  auto store_tiles = [&]() {
#pragma unroll
    for (int it = 0; it < 4; ++it) {
      const int idx = t + (it << 8);
      if constexpr (AKC) {
        const int row = idx >> 3, k4 = (idx & 7) << 2;
        h4 hv = {(_Float16)pa[it].x, (_Float16)pa[it].y, (_Float16)pa[it].z, (_Float16)pa[it].w};
        *(h4*)&As[row * LDP + k4] = hv;
      } else {
        const int col = idx >> 5, r4 = (idx & 31) << 2;
        As[(r4 + 0) * LDP + col] = (_Float16)pa[it].x;
        As[(r4 + 1) * LDP + col] = (_Float16)pa[it].y;
        As[(r4 + 2) * LDP + col] = (_Float16)pa[it].z;
        As[(r4 + 3) * LDP + col] = (_Float16)pa[it].w;
      }
    }
#pragma unroll
    for (int it = 0; it < 2; ++it) {
      const int idx = t + (it << 8);
      if constexpr (BKC) {
        const int row = idx >> 3, k4 = (idx & 7) << 2;
        h4 hv = {(_Float16)pb[it].x, (_Float16)pb[it].y, (_Float16)pb[it].z, (_Float16)pb[it].w};
        *(h4*)&Bs[row * LDP + k4] = hv;
      } else {
        const int col = idx >> 4, n4 = (idx & 15) << 2;
        Bs[(n4 + 0) * LDP + col] = (_Float16)pb[it].x;
        Bs[(n4 + 1) * LDP + col] = (_Float16)pb[it].y;
        Bs[(n4 + 2) * LDP + col] = (_Float16)pb[it].z;
        Bs[(n4 + 3) * LDP + col] = (_Float16)pb[it].w;
      }
    }
  };

  v8f acc[2][2] = {};

  load_tiles(0);
  for (int k0 = 0; k0 < K; k0 += BK) {
    store_tiles();
    __syncthreads();
    if (k0 + BK < K) load_tiles(k0 + BK);   // prefetch next tile while WMMAs run

    v16h a0 = frag_a(As, wm,      lane);
    v16h a1 = frag_a(As, wm + 16, lane);
    v16h b0 = frag_b(Bs, wn,      lane);
    v16h b1 = frag_b(Bs, wn + 16, lane);
    acc[0][0] = __builtin_amdgcn_wmma_f32_16x16x32_f16(false, a0, false, b0, (short)0, acc[0][0], false, false);
    acc[0][1] = __builtin_amdgcn_wmma_f32_16x16x32_f16(false, a0, false, b1, (short)0, acc[0][1], false, false);
    acc[1][0] = __builtin_amdgcn_wmma_f32_16x16x32_f16(false, a1, false, b0, (short)0, acc[1][0], false, false);
    acc[1][1] = __builtin_amdgcn_wmma_f32_16x16x32_f16(false, a1, false, b1, (short)0, acc[1][1], false, false);
    __syncthreads();
  }

  // C/D layout: VGPR j -> M = j + 8*(lane>=16), N = lane&15
  const int ln = lane & 15;
  const int mh = (lane >> 4) << 3;
#pragma unroll
  for (int mi = 0; mi < 2; ++mi)
#pragma unroll
    for (int ni = 0; ni < 2; ++ni) {
      const int gn = bn + wn + ni * 16 + ln;
      const float bs = bias ? bias[gn] : 0.f;
#pragma unroll
      for (int j = 0; j < 8; ++j) {
        const int gm = bm + wm + mi * 16 + mh + j;
        const float v = acc[mi][ni][j] + bs;
        const long long o = z * d_batch + (long long)gm * N + gn;
        if constexpr (OUTH) Dh[o] = (_Float16)v; else Df[o] = v;
      }
    }
}

// bc_l[o] = b_val[o] + sum_k W_val[o,k]*(bv_l[k] + lvl_emb[l,k])
__global__ void bias_combine_levels(const float* __restrict__ W_val, const float* __restrict__ b_val,
                                    const float* __restrict__ bv2, const float* __restrict__ bv3,
                                    const float* __restrict__ bv4, const float* __restrict__ bv5,
                                    const float* __restrict__ lvl_emb, float* __restrict__ bc) {
  int l = blockIdx.x, o = threadIdx.x;
  const float* bv = (l == 0) ? bv2 : (l == 1) ? bv3 : (l == 2) ? bv4 : bv5;
  float s = b_val[o];
  const float* wrow = W_val + o * 256;
  const float* er = lvl_emb + l * 256;
  for (int k = 0; k < 256; k += 4) {
    float4 w = *(const float4*)(wrow + k);
    float4 b = *(const float4*)(bv + k);
    float4 e = *(const float4*)(er + k);
    s += w.x * (b.x + e.x) + w.y * (b.y + e.y) + w.z * (b.z + e.z) + w.w * (b.w + e.w);
  }
  bc[l * 256 + o] = s;
}

// bco[o] = b_out[o] + sum_j W_out[o,j]*b_oa[j]
__global__ void bias_combine_out(const float* __restrict__ W_out, const float* __restrict__ b_out,
                                 const float* __restrict__ b_oa, float* __restrict__ bco) {
  int o = blockIdx.x * 256 + threadIdx.x;
  float s = b_out[o];
  const float* wrow = W_out + o * 256;
  for (int j = 0; j < 256; j += 4) {
    float4 w = *(const float4*)(wrow + j);
    float4 b = *(const float4*)(b_oa + j);
    s += w.x * b.x + w.y * b.y + w.z * b.z + w.w * b.w;
  }
  bco[o] = s;
}

// cosine-sim argmax over res5 positions -> reference point (x,y) in [0,1]
__global__ void refpoint_kernel(const float* __restrict__ q, const float* __restrict__ vals5,
                                float* __restrict__ refbuf) {
  __shared__ float qrow[256];
  __shared__ float sc[256];
  __shared__ int   si[256];
  int bq = blockIdx.x, b = bq >> 8, t = threadIdx.x;
  qrow[t] = q[(long long)bq * 256 + t];
  __syncthreads();
  float best = -1e30f; int bi = 0;
  for (int l = t; l < 1024; l += 256) {
    const float* f = vals5 + ((long long)b * 1024 + l) * 256;
    float dot = 0.f, nsq = 0.f;
    for (int c = 0; c < 256; c += 4) {
      float4 fv = *(const float4*)(f + c);
      dot += qrow[c] * fv.x + qrow[c + 1] * fv.y + qrow[c + 2] * fv.z + qrow[c + 3] * fv.w;
      nsq += fv.x * fv.x + fv.y * fv.y + fv.z * fv.z + fv.w * fv.w;
    }
    float s = dot / fmaxf(sqrtf(nsq), 1e-12f);  // ||q|| is a common positive scale -> argmax invariant
    if (s > best || (s == best && l < bi)) { best = s; bi = l; }
  }
  sc[t] = best; si[t] = bi;
  __syncthreads();
  for (int stp = 128; stp > 0; stp >>= 1) {
    if (t < stp) {
      float s2 = sc[t + stp]; int i2 = si[t + stp];
      if (s2 > sc[t] || (s2 == sc[t] && i2 < si[t])) { sc[t] = s2; si[t] = i2; }
    }
    __syncthreads();
  }
  if (t == 0) {
    int idx = si[0];
    refbuf[bq * 2 + 0] = ((idx & 31) + 0.5f) / 32.f;  // W5 = 32
    refbuf[bq * 2 + 1] = ((idx >> 5) + 0.5f) / 32.f;  // H5 = 32
  }
}

// offsets (256) + attention logits (128) + per-head softmax over 16
__global__ void offaw_kernel(const float* __restrict__ q,
                             const float* __restrict__ W_off, const float* __restrict__ b_off,
                             const float* __restrict__ W_aw,  const float* __restrict__ b_aw,
                             float* __restrict__ offbuf, float* __restrict__ awbuf) {
  __shared__ float qrow[256];
  __shared__ float awrow[128];
  int bq = blockIdx.x, t = threadIdx.x;
  qrow[t] = q[(long long)bq * 256 + t];
  __syncthreads();
  {
    float s = b_off[t];
    const float* wrow = W_off + t * 256;
    for (int k = 0; k < 256; k += 4) {
      float4 w = *(const float4*)(wrow + k);
      s += qrow[k] * w.x + qrow[k + 1] * w.y + qrow[k + 2] * w.z + qrow[k + 3] * w.w;
    }
    offbuf[(long long)bq * 256 + t] = s;
  }
  if (t < 128) {
    float a = b_aw[t];
    const float* wrow = W_aw + t * 256;
    for (int k = 0; k < 256; k += 4) {
      float4 w = *(const float4*)(wrow + k);
      a += qrow[k] * w.x + qrow[k + 1] * w.y + qrow[k + 2] * w.z + qrow[k + 3] * w.w;
    }
    awrow[t] = a;
  }
  __syncthreads();
  if (t < 8) {  // one thread per head: softmax over NL*NP = 16
    float m = -1e30f;
    for (int j = 0; j < 16; ++j) m = fmaxf(m, awrow[t * 16 + j]);
    float e[16], sum = 0.f;
    for (int j = 0; j < 16; ++j) { e[j] = __expf(awrow[t * 16 + j] - m); sum += e[j]; }
    for (int j = 0; j < 16; ++j) awbuf[(long long)bq * 128 + t * 16 + j] = e[j] / sum;
  }
}

// bilinear sampling + attention-weighted sum. wave = head, lane = dh channel.
__global__ void sample_kernel(const _Float16* __restrict__ vbuf, const float* __restrict__ refbuf,
                              const float* __restrict__ offbuf, const float* __restrict__ awbuf,
                              float* __restrict__ out1) {
  const int Hs[4]   = {256, 128, 64, 32};
  const int Ws[4]   = {256, 128, 64, 32};
  const int offL[4] = {0, 65536, 81920, 86016};
  const long long LTOT = 87040;
  int bq = blockIdx.x, b = bq >> 8;
  int t = threadIdx.x, h = t >> 5;
  float rx = refbuf[bq * 2 + 0], ry = refbuf[bq * 2 + 1];
  float acc = 0.f;
  for (int l = 0; l < 4; ++l) {
    const int H = Hs[l], W = Ws[l];
    const _Float16* vl = vbuf + ((long long)b * LTOT + offL[l]) * 256;
    for (int p = 0; p < 4; ++p) {
      int pi = (h * 4 + l) * 4 + p;
      float ox = offbuf[(long long)bq * 256 + pi * 2 + 0];
      float oy = offbuf[(long long)bq * 256 + pi * 2 + 1];
      float aw = awbuf[(long long)bq * 128 + pi];
      float x = (rx + ox / (float)W) * W - 0.5f;
      float y = (ry + oy / (float)H) * H - 0.5f;
      float xf = floorf(x), yf = floorf(y);
      int x0 = (int)xf, y0 = (int)yf;
      float wx = x - xf, wy = y - yf;
      float v00 = 0.f, v01 = 0.f, v10 = 0.f, v11 = 0.f;
      if (x0 >= 0     && x0 < W     && y0 >= 0     && y0 < H)     v00 = (float)vl[((long long)(y0 * W + x0)) * 256 + t];
      if (x0 + 1 >= 0 && x0 + 1 < W && y0 >= 0     && y0 < H)     v01 = (float)vl[((long long)(y0 * W + x0 + 1)) * 256 + t];
      if (x0 >= 0     && x0 < W     && y0 + 1 >= 0 && y0 + 1 < H) v10 = (float)vl[((long long)((y0 + 1) * W + x0)) * 256 + t];
      if (x0 + 1 >= 0 && x0 + 1 < W && y0 + 1 >= 0 && y0 + 1 < H) v11 = (float)vl[((long long)((y0 + 1) * W + x0 + 1)) * 256 + t];
      acc += aw * ((v00 * (1.f - wx) + v01 * wx) * (1.f - wy) + (v10 * (1.f - wx) + v11 * wx) * wy);
    }
  }
  out1[(long long)bq * 256 + t] = acc;
}

extern "C" void kernel_launch(void* const* d_in, const int* in_sizes, int n_in,
                              void* d_out, int out_size, void* d_ws, size_t ws_size,
                              hipStream_t stream) {
  (void)in_sizes; (void)n_in; (void)out_size; (void)ws_size;
  const float* queries = (const float*)d_in[0];
  const float* res[4]  = {(const float*)d_in[1], (const float*)d_in[2], (const float*)d_in[3], (const float*)d_in[4]};
  const float* Wq      = (const float*)d_in[5];
  const float* bq      = (const float*)d_in[6];
  const float* Wv[4]   = {(const float*)d_in[7], (const float*)d_in[9], (const float*)d_in[11], (const float*)d_in[13]};
  const float* bv[4]   = {(const float*)d_in[8], (const float*)d_in[10], (const float*)d_in[12], (const float*)d_in[14]};
  const float* lvl_emb = (const float*)d_in[15];
  const float* W_val   = (const float*)d_in[16];
  const float* b_val   = (const float*)d_in[17];
  const float* W_off   = (const float*)d_in[18];
  const float* b_off   = (const float*)d_in[19];
  const float* W_aw    = (const float*)d_in[20];
  const float* b_aw    = (const float*)d_in[21];
  const float* W_oa    = (const float*)d_in[22];
  const float* b_oa    = (const float*)d_in[23];
  const float* W_out   = (const float*)d_in[24];
  const float* b_out   = (const float*)d_in[25];
  float* out = (float*)d_out;

  const int Cs[4]  = {128, 256, 512, 1024};
  const int HWs[4] = {65536, 16384, 4096, 1024};
  const int offL[4] = {0, 65536, 81920, 86016};
  const long long LTOT = 87040;

  char* ws = (char*)d_ws;
  size_t woff = 0;
  auto alloc = [&](size_t bytes) { void* p = ws + woff; woff = (woff + bytes + 255) & ~(size_t)255; return p; };

  float* wc[4];
  for (int l = 0; l < 4; ++l) wc[l] = (float*)alloc((size_t)256 * Cs[l] * 4);
  float*    bc     = (float*)alloc(4 * 256 * 4);
  float*    wcout  = (float*)alloc((size_t)2560 * 256 * 4);
  float*    bcout  = (float*)alloc(2560 * 4);
  float*    qbuf   = (float*)alloc((size_t)1024 * 256 * 4);
  float*    vals5  = (float*)alloc((size_t)4 * 1024 * 256 * 4);
  _Float16* vbuf   = (_Float16*)alloc((size_t)4 * LTOT * 256 * 2);
  float*    refbuf = (float*)alloc(1024 * 2 * 4);
  float*    offbuf = (float*)alloc((size_t)1024 * 256 * 4);
  float*    awbuf  = (float*)alloc((size_t)1024 * 128 * 4);
  float*    out1   = (float*)alloc((size_t)1024 * 256 * 4);

  dim3 blk(256);

  // 1) combined weights Wc_l[o,c] = sum_k W_val[o,k]*Wv_l[k,c]   (A k-contig, B n-contig)
  for (int l = 0; l < 4; ++l) {
    dim3 g(256 / BM, Cs[l] / BN, 1);
    gemm_wmma_f16<true, false, false><<<g, blk, 0, stream>>>(
        W_val, 256, 1, 0, Wv[l], 1, Cs[l], nullptr, wc[l], nullptr, 0, Cs[l], 256);
  }
  bias_combine_levels<<<4, 256, 0, stream>>>(W_val, b_val, bv[0], bv[1], bv[2], bv[3], lvl_emb, bc);

  // 2) Wc_out[o,k] = sum_j W_out[o,j]*W_oa[j,k]   (A k-contig, B n-contig)
  {
    dim3 g(2560 / BM, 256 / BN, 1);
    gemm_wmma_f16<true, false, false><<<g, blk, 0, stream>>>(
        W_out, 256, 1, 0, W_oa, 1, 256, nullptr, wcout, nullptr, 0, 256, 256);
  }
  bias_combine_out<<<10, 256, 0, stream>>>(W_out, b_out, b_oa, bcout);

  // 3) q = queries @ Wq^T + bq   (A k-contig, B k-contig)
  {
    dim3 g(1024 / BM, 256 / BN, 1);
    gemm_wmma_f16<true, true, false><<<g, blk, 0, stream>>>(
        queries, 2560, 1, 0, Wq, 2560, 1, bq, qbuf, nullptr, 0, 256, 2560);
  }

  // 4) vals5 = res5 @ Wv5^T + bv5 (per batch)   (A m-contig, B k-contig)
  {
    dim3 g(1024 / BM, 256 / BN, 4);
    gemm_wmma_f16<false, true, false><<<g, blk, 0, stream>>>(
        res[3], 1, 1024, (long long)1024 * 1024, Wv[3], 1024, 1, bv[3],
        vals5, nullptr, (long long)1024 * 256, 256, 1024);
  }

  // 5) fused v_l = res_l @ Wc_l^T + bc_l  -> f16 value buffer (B, LTOT, 256)
  for (int l = 0; l < 4; ++l) {
    dim3 g(HWs[l] / BM, 256 / BN, 4);
    gemm_wmma_f16<false, true, true><<<g, blk, 0, stream>>>(
        res[l], 1, HWs[l], (long long)Cs[l] * HWs[l], wc[l], Cs[l], 1, bc + l * 256,
        nullptr, vbuf + (long long)offL[l] * 256, LTOT * 256, 256, Cs[l]);
  }

  // 6) reference points, offsets/weights, sampling
  refpoint_kernel<<<1024, 256, 0, stream>>>(qbuf, vals5, refbuf);
  offaw_kernel<<<1024, 256, 0, stream>>>(qbuf, W_off, b_off, W_aw, b_aw, offbuf, awbuf);
  sample_kernel<<<1024, 256, 0, stream>>>(vbuf, refbuf, offbuf, awbuf, out1);

  // 7) out = out1 @ Wc_out^T + bc_out   (A k-contig, B k-contig)
  {
    dim3 g(1024 / BM, 2560 / BN, 1);
    gemm_wmma_f16<true, true, false><<<g, blk, 0, stream>>>(
        out1, 256, 1, 0, wcout, 256, 1, bcout, out, nullptr, 0, 2560, 256);
  }
}
